// InducedSetAttention_17171279250124
// MI455X (gfx1250) — compile-verified, matrix operand order checked by reading, not run on hardware
//
#include <hip/hip_runtime.h>
#include <hip/hip_bf16.h>

#define B_   16
#define L_   4096
#define D_   512
#define H_   8
#define NI_  64
#define DQK_ 64
#define BL_  (B_*L_)
#define NSEG 8
#define SEGL (L_/NSEG)   // 512 keys per attn0 segment

typedef __attribute__((ext_vector_type(16))) __bf16 v16bf;
typedef __attribute__((ext_vector_type(8)))  __bf16 v8bf;
typedef __attribute__((ext_vector_type(8)))  float  v8f;

__device__ __forceinline__ v8f vzero8() {
  v8f z;
#pragma unroll
  for (int i = 0; i < 8; ++i) z[i] = 0.f;
  return z;
}

__device__ __forceinline__ v8f wmma_bf16(v16bf a, v16bf b, v8f c) {
  // D = A(16x32 bf16) x B(32x16 bf16) + C(16x16 f32)
  return __builtin_amdgcn_wmma_f32_16x16x32_bf16(false, a, false, b, (short)0, c,
                                                 false, false);
}

// relu(x)^1.5 with fast hardware sqrt (bf16 pipeline, precision is ample)
__device__ __forceinline__ float relu15(float x) {
  x = fmaxf(x, 0.f);
  return x * __builtin_amdgcn_sqrtf(x);
}

// Load a 16x32 bf16 WMMA operand fragment.  "row" maps to lanes (A: row==M,
// B: row==N) and K runs contiguously in memory.  Per ISA layout, half h of
// lane L holds K = k0 + (h&7) + 8*(L>>4) + 16*(h>>3): two contiguous 8-element
// (16B) runs per lane -> two b128 loads.
__device__ __forceinline__ v16bf load_frag(const __bf16* base, int row, int ld, int k0) {
  const int lane = threadIdx.x & 31;
  const int hs   = lane >> 4;
  const __bf16* p = base + (size_t)row * ld + k0 + 8 * hs;
  v8bf lo = *(const v8bf*)(p);
  v8bf hi = *(const v8bf*)(p + 16);
  v16bf r;
#pragma unroll
  for (int i = 0; i < 8; ++i) { r[i] = lo[i]; r[i + 8] = hi[i]; }
  return r;
}

// ---------------------------------------------------------------------------
// Generic bf16 GEMM:  C[M,N] = A[M,K] * W[N,K]^T   (i.e. x @ W.T)
// Block: 256 threads = 8 waves (2x4), block tile 64x256, wave tile 32x64.
// MODE 0: f32 out      MODE 1: bf16 out
// MODE 2: bf16 transposed-out (per-batch [N, L] layout, for V^T)
// MODE 3: f32 out + residual add
// ---------------------------------------------------------------------------
template <int MODE>
__global__ __launch_bounds__(256, 1)
void gemm_k(const __bf16* __restrict__ A, const __bf16* __restrict__ W,
            void* __restrict__ Cout, const float* __restrict__ resid,
            int M, int N, int K, int lda, int ldw, int ldc) {
  const int wid  = threadIdx.x >> 5;
  const int lane = threadIdx.x & 31;
  const int r16  = lane & 15;
  const int hf   = lane >> 4;
  const int m0 = blockIdx.x * 64 + (wid >> 2) * 32;
  const int n0 = blockIdx.y * 256 + (wid & 3) * 64;

  v8f acc[2][4];
#pragma unroll
  for (int i = 0; i < 2; ++i)
#pragma unroll
    for (int j = 0; j < 4; ++j) acc[i][j] = vzero8();

  for (int k0 = 0; k0 < K; k0 += 32) {
    v16bf af[2], bf[4];
#pragma unroll
    for (int i = 0; i < 2; ++i) af[i] = load_frag(A, m0 + i * 16 + r16, lda, k0);
#pragma unroll
    for (int j = 0; j < 4; ++j) bf[j] = load_frag(W, n0 + j * 16 + r16, ldw, k0);
    if (k0 + 64 < K) {
      // prefetch next-next k-step of the streaming A operand (global_prefetch_b8)
      __builtin_prefetch(A + (size_t)(m0 + r16) * lda + k0 + 64, 0, 1);
      __builtin_prefetch(A + (size_t)(m0 + 16 + r16) * lda + k0 + 64, 0, 1);
    }
#pragma unroll
    for (int i = 0; i < 2; ++i)
#pragma unroll
      for (int j = 0; j < 4; ++j) acc[i][j] = wmma_bf16(af[i], bf[j], acc[i][j]);
  }

#pragma unroll
  for (int i = 0; i < 2; ++i)
#pragma unroll
    for (int j = 0; j < 4; ++j) {
      if (MODE == 2) {
        // transposed store: out[b][n][l], consecutive r -> consecutive l
        const int mstart = m0 + i * 16 + 8 * hf;
        const int bb = mstart >> 12;        // / L_
        const int l  = mstart & (L_ - 1);
        const int n  = n0 + j * 16 + r16;
        __bf16* dst = (__bf16*)Cout + (size_t)bb * N * L_ + (size_t)n * L_ + l;
        v8bf pk;
#pragma unroll
        for (int r = 0; r < 8; ++r) pk[r] = (__bf16)acc[i][j][r];
        *(v8bf*)dst = pk;
      } else {
#pragma unroll
        for (int r = 0; r < 8; ++r) {
          const int m = m0 + i * 16 + r + 8 * hf;
          const int n = n0 + j * 16 + r16;
          float vv = acc[i][j][r];
          if (MODE == 3) vv += resid[(size_t)m * ldc + n];
          if (MODE == 1) ((__bf16*)Cout)[(size_t)m * ldc + n] = (__bf16)vv;
          else           ((float*)Cout)[(size_t)m * ldc + n] = vv;
        }
      }
    }
}

// ---------------------------------------------------------------------------
// attn0: per (b, h, key-segment) workgroup.  S = (I/T) @ ks^T  [64 x 512],
// w = relu(S)^1.5, partial rowsum and partial Hm_num = w @ vs  [64 x 64].
// ---------------------------------------------------------------------------
__global__ __launch_bounds__(256, 1)
void attn0_k(const __bf16* __restrict__ Ibf,   // [64, 512] pre-scaled 1/T
             const __bf16* __restrict__ ks,    // [B*L, 512]
             const __bf16* __restrict__ vsT,   // [B, 512, L]
             float* __restrict__ pHm,          // [B,H,NSEG,64,64]
             float* __restrict__ pRs) {        // [B,H,NSEG,64]
  const int bid = blockIdx.x;
  const int seg = bid & (NSEG - 1);
  const int h   = (bid >> 3) & 7;
  const int b   = bid >> 6;
  const int wid  = threadIdx.x >> 5;
  const int r16  = threadIdx.x & 15;
  const int hf   = (threadIdx.x & 31) >> 4;

  __shared__ __bf16 wl[8][64][72];   // per-wave relu15 weight tile
  __shared__ float  hm[64][64];
  __shared__ float  rs[64];
  for (int t = threadIdx.x; t < 64 * 64; t += 256) (&hm[0][0])[t] = 0.f;
  if (threadIdx.x < 64) rs[threadIdx.x] = 0.f;
  __syncthreads();

  const int lbase = seg * SEGL + wid * 64;   // this wave's 64 keys
  const __bf16* ksb = ks + (size_t)b * L_ * D_;
  const __bf16* vsb = vsT + (size_t)b * D_ * L_;

  // scores: M = 64 induced pts (4 frags), N = 64 keys (4 frags), K = 64 (d)
  v8f acc[4][4];
#pragma unroll
  for (int i = 0; i < 4; ++i)
#pragma unroll
    for (int j = 0; j < 4; ++j) acc[i][j] = vzero8();
#pragma unroll 1
  for (int k0 = 0; k0 < 64; k0 += 32) {
    v16bf af[4], bf[4];
#pragma unroll
    for (int i = 0; i < 4; ++i) af[i] = load_frag(Ibf, i * 16 + r16, D_, h * 64 + k0);
#pragma unroll
    for (int j = 0; j < 4; ++j)
      bf[j] = load_frag(ksb, lbase + j * 16 + r16, D_, h * 64 + k0);
#pragma unroll
    for (int i = 0; i < 4; ++i)
#pragma unroll
      for (int j = 0; j < 4; ++j) acc[i][j] = wmma_bf16(af[i], bf[j], acc[i][j]);
  }

  // w = relu(S)^1.5, accumulate rowsums, spill w tile to LDS (bf16)
#pragma unroll
  for (int i = 0; i < 4; ++i)
#pragma unroll
    for (int r = 0; r < 8; ++r) {
      float p = 0.f;
      const int m = i * 16 + r + 8 * hf;
#pragma unroll
      for (int j = 0; j < 4; ++j) {
        float w = relu15(acc[i][j][r]);
        p += w;
        wl[wid][m][j * 16 + r16] = (__bf16)w;
      }
      p += __shfl_xor(p, 1, 32); p += __shfl_xor(p, 2, 32);
      p += __shfl_xor(p, 4, 32); p += __shfl_xor(p, 8, 32);
      if (r16 == 0) atomicAdd(&rs[m], p);
    }

  // Hm_num partial: M = 64 i (4 frags), N = 64 dv (4 frags), K = 64 local keys
  v8f vac[4][4];
#pragma unroll
  for (int i = 0; i < 4; ++i)
#pragma unroll
    for (int j = 0; j < 4; ++j) vac[i][j] = vzero8();
#pragma unroll 1
  for (int k0 = 0; k0 < 64; k0 += 32) {
    v16bf af[4], bf[4];
#pragma unroll
    for (int i = 0; i < 4; ++i) af[i] = load_frag(&wl[wid][0][0], i * 16 + r16, 72, k0);
#pragma unroll
    for (int j = 0; j < 4; ++j)
      bf[j] = load_frag(vsb, h * 64 + j * 16 + r16, L_, lbase + k0);
#pragma unroll
    for (int i = 0; i < 4; ++i)
#pragma unroll
      for (int j = 0; j < 4; ++j) vac[i][j] = wmma_bf16(af[i], bf[j], vac[i][j]);
  }
#pragma unroll
  for (int i = 0; i < 4; ++i)
#pragma unroll
    for (int j = 0; j < 4; ++j)
#pragma unroll
      for (int r = 0; r < 8; ++r)
        atomicAdd(&hm[i * 16 + r + 8 * hf][j * 16 + r16], vac[i][j][r]);
  __syncthreads();

  float* pH = pHm + (size_t)((b * 8 + h) * NSEG + seg) * 64 * 64;
  for (int t = threadIdx.x; t < 64 * 64; t += 256) pH[t] = (&hm[0][0])[t];
  float* pR = pRs + ((b * 8 + h) * NSEG + seg) * 64;
  if (threadIdx.x < 64) pR[threadIdx.x] = rs[threadIdx.x];
}

// Reduce attn0 partials over segments, normalize, emit merged-head bf16 [B,64,512]
__global__ void reduce0_k(const float* __restrict__ pHm, const float* __restrict__ pRs,
                          __bf16* __restrict__ Hm0) {
  const int bh = blockIdx.x;          // B*H
  const int b = bh >> 3, h = bh & 7;
  for (int t = threadIdx.x; t < NI_ * DQK_; t += 256) {
    const int i = t >> 6, dv = t & 63;
    float s = 0.f, rsum = 0.f;
    for (int seg = 0; seg < NSEG; ++seg) {
      s    += pHm[(size_t)(bh * NSEG + seg) * 4096 + t];
      rsum += pRs[(bh * NSEG + seg) * 64 + i];
    }
    Hm0[(size_t)(b * NI_ + i) * D_ + h * 64 + dv] =
        (__bf16)(s * __builtin_amdgcn_rcpf(rsum + 1e-6f));
  }
}

// Split HmW [B,64,512] f32 into per-head row-major + transposed bf16 copies
__global__ void hmsplit_k(const float* __restrict__ HmW,
                          __bf16* __restrict__ Hm, __bf16* __restrict__ HmT) {
  const int t = blockIdx.x * 256 + threadIdx.x;   // B*64*512 = 524288
  const int c = t & 511;
  const int i = (t >> 9) & 63;
  const int b = t >> 15;
  const int h = c >> 6, d = c & 63;
  const float vv = HmW[t];
  Hm [(size_t)((b * H_ + h) * NI_ + i) * DQK_ + d] = (__bf16)vv;
  HmT[(size_t)((b * H_ + h) * DQK_ + d) * NI_ + i] = (__bf16)vv;
}

// ---------------------------------------------------------------------------
// attn1: WG = (b, 64-query tile); wave = head.  S1 = (qs/T) @ Hm^T [64 x 64],
// w = relu^1.5, O = (w @ Hm) / (rowsum + 1e-6)  -> bf16 [B*L, 512]
// ---------------------------------------------------------------------------
__global__ __launch_bounds__(256, 1)
void attn1_k(const __bf16* __restrict__ qs,  // [B*L, 512]
             const __bf16* __restrict__ Hm,  // [B,H,64,64]
             const __bf16* __restrict__ HmT, // [B,H,64,64] transposed
             __bf16* __restrict__ O) {       // [B*L, 512]
  const int bid = blockIdx.x;           // B * (L/64)
  const int b  = bid >> 6;
  const int q0 = (bid & 63) * 64;
  const int h  = threadIdx.x >> 5;      // wave == head
  const int r16 = threadIdx.x & 15;
  const int hf  = (threadIdx.x & 31) >> 4;
  __shared__ __bf16 wl[8][64][72];

  const __bf16* qsb = qs + ((size_t)b * L_ + q0) * D_;
  const __bf16* hmb = Hm  + (size_t)(b * H_ + h) * NI_ * DQK_;
  const __bf16* hmt = HmT + (size_t)(b * H_ + h) * NI_ * DQK_;

  v8f acc[4][4];
#pragma unroll
  for (int i = 0; i < 4; ++i)
#pragma unroll
    for (int j = 0; j < 4; ++j) acc[i][j] = vzero8();
#pragma unroll 1
  for (int k0 = 0; k0 < 64; k0 += 32) {
    v16bf af[4], bf[4];
#pragma unroll
    for (int mf = 0; mf < 4; ++mf)
      af[mf] = load_frag(qsb, mf * 16 + r16, D_, h * 64 + k0);
#pragma unroll
    for (int j = 0; j < 4; ++j) bf[j] = load_frag(hmb, j * 16 + r16, DQK_, k0);
#pragma unroll
    for (int mf = 0; mf < 4; ++mf)
#pragma unroll
      for (int j = 0; j < 4; ++j) acc[mf][j] = wmma_bf16(af[mf], bf[j], acc[mf][j]);
  }

  float rinv[4][8];
#pragma unroll
  for (int mf = 0; mf < 4; ++mf)
#pragma unroll
    for (int r = 0; r < 8; ++r) {
      float p = 0.f;
      const int m = mf * 16 + r + 8 * hf;
#pragma unroll
      for (int j = 0; j < 4; ++j) {
        float w = relu15(acc[mf][j][r] * 0.125f);   // 1/TEMP folded in
        p += w;
        wl[h][m][j * 16 + r16] = (__bf16)w;
      }
      p += __shfl_xor(p, 1, 32); p += __shfl_xor(p, 2, 32);
      p += __shfl_xor(p, 4, 32); p += __shfl_xor(p, 8, 32);
      // lane's half holds exactly its row's sum; keep reciprocal for epilogue
      rinv[mf][r] = __builtin_amdgcn_rcpf(p + 1e-6f);
    }

  v8f oac[4][4];
#pragma unroll
  for (int i = 0; i < 4; ++i)
#pragma unroll
    for (int j = 0; j < 4; ++j) oac[i][j] = vzero8();
#pragma unroll 1
  for (int k0 = 0; k0 < 64; k0 += 32) {
    v16bf af[4], bf[4];
#pragma unroll
    for (int mf = 0; mf < 4; ++mf)
      af[mf] = load_frag(&wl[h][0][0], mf * 16 + r16, 72, k0);
#pragma unroll
    for (int j = 0; j < 4; ++j) bf[j] = load_frag(hmt, j * 16 + r16, DQK_, k0);
#pragma unroll
    for (int mf = 0; mf < 4; ++mf)
#pragma unroll
      for (int j = 0; j < 4; ++j) oac[mf][j] = wmma_bf16(af[mf], bf[j], oac[mf][j]);
  }
#pragma unroll
  for (int mf = 0; mf < 4; ++mf)
#pragma unroll
    for (int j = 0; j < 4; ++j)
#pragma unroll
      for (int r = 0; r < 8; ++r) {
        const int m = mf * 16 + r + 8 * hf;
        O[((size_t)b * L_ + q0 + m) * D_ + h * 64 + j * 16 + r16] =
            (__bf16)(oac[mf][j][r] * rinv[mf][r]);
      }
}

__global__ void cvt_k(const float* __restrict__ src, __bf16* __restrict__ dst,
                      int n, float scale) {
  for (int i = blockIdx.x * blockDim.x + threadIdx.x; i < n;
       i += gridDim.x * blockDim.x)
    dst[i] = (__bf16)(src[i] * scale);
}

__global__ void ln_k(float* __restrict__ Y, const float* __restrict__ lnw,
                     const float* __restrict__ lnb) {
  __shared__ float red[256];
  const int t = threadIdx.x;
  float* p = Y + (size_t)blockIdx.x * D_;
  const float x0 = p[t], x1 = p[t + 256];
  red[t] = x0 + x1;
  __syncthreads();
  for (int off = 128; off > 0; off >>= 1) {
    if (t < off) red[t] += red[t + off];
    __syncthreads();
  }
  const float mu = red[0] * (1.f / D_);
  __syncthreads();
  const float d0 = x0 - mu, d1 = x1 - mu;
  red[t] = d0 * d0 + d1 * d1;
  __syncthreads();
  for (int off = 128; off > 0; off >>= 1) {
    if (t < off) red[t] += red[t + off];
    __syncthreads();
  }
  const float inv = rsqrtf(red[0] * (1.f / D_) + 1e-6f);
  p[t]       = d0 * inv * lnw[t] + lnb[t];
  p[t + 256] = d1 * inv * lnw[t + 256] + lnb[t + 256];
}

extern "C" void kernel_launch(void* const* d_in, const int* in_sizes, int n_in,
                              void* d_out, int out_size, void* d_ws, size_t ws_size,
                              hipStream_t stream) {
  const float* q   = (const float*)d_in[0];
  const float* k   = (const float*)d_in[1];
  const float* v   = (const float*)d_in[2];
  const float* I   = (const float*)d_in[3];
  const float* Wq  = (const float*)d_in[4];
  const float* Wk  = (const float*)d_in[5];
  const float* Wv  = (const float*)d_in[6];
  const float* Ww  = (const float*)d_in[7];
  const float* Wfc = (const float*)d_in[8];
  const float* lnw = (const float*)d_in[9];
  const float* lnb = (const float*)d_in[10];

  char* base = (char*)d_ws;
  size_t off = 0;
  auto alloc = [&](size_t bytes) -> void* {
    void* p = base + off;
    off += (bytes + 255) & ~(size_t)255;
    return p;
  };
  __bf16* Wqb  = (__bf16*)alloc((size_t)D_ * D_ * 2);
  __bf16* Wkb  = (__bf16*)alloc((size_t)D_ * D_ * 2);
  __bf16* Wvb  = (__bf16*)alloc((size_t)D_ * D_ * 2);
  __bf16* Wwb  = (__bf16*)alloc((size_t)D_ * D_ * 2);
  __bf16* Wfcb = (__bf16*)alloc((size_t)D_ * D_ * 2);
  __bf16* Ibf  = (__bf16*)alloc((size_t)NI_ * D_ * 2);
  __bf16* qsb  = (__bf16*)alloc((size_t)BL_ * D_ * 2);
  __bf16* ksb  = (__bf16*)alloc((size_t)BL_ * D_ * 2);
  __bf16* vst  = (__bf16*)alloc((size_t)BL_ * D_ * 2);
  __bf16* tmp  = (__bf16*)alloc((size_t)BL_ * D_ * 2);  // bf16 input, later O
  float*  pHm  = (float*)alloc((size_t)B_ * H_ * NSEG * 64 * 64 * 4);
  float*  pRs  = (float*)alloc((size_t)B_ * H_ * NSEG * 64 * 4);
  __bf16* Hm0  = (__bf16*)alloc((size_t)B_ * NI_ * D_ * 2);
  float*  HmW  = (float*)alloc((size_t)B_ * NI_ * D_ * 4);
  __bf16* Hmb  = (__bf16*)alloc((size_t)B_ * H_ * NI_ * DQK_ * 2);
  __bf16* HmT  = (__bf16*)alloc((size_t)B_ * H_ * NI_ * DQK_ * 2);

  // 1. weight / induced-point conversions (I pre-scaled by 1/TEMP = 0.125)
  cvt_k<<<512, 256, 0, stream>>>(Wq,  Wqb,  D_ * D_, 1.f);
  cvt_k<<<512, 256, 0, stream>>>(Wk,  Wkb,  D_ * D_, 1.f);
  cvt_k<<<512, 256, 0, stream>>>(Wv,  Wvb,  D_ * D_, 1.f);
  cvt_k<<<512, 256, 0, stream>>>(Ww,  Wwb,  D_ * D_, 1.f);
  cvt_k<<<512, 256, 0, stream>>>(Wfc, Wfcb, D_ * D_, 1.f);
  cvt_k<<<128, 256, 0, stream>>>(I,   Ibf,  NI_ * D_, 0.125f);

  const dim3 gP(BL_ / 64, D_ / 256);
  // 2. projections (reuse `tmp` as bf16 staging per tensor)
  cvt_k<<<8192, 256, 0, stream>>>(q, tmp, BL_ * D_, 1.f);
  gemm_k<1><<<gP, 256, 0, stream>>>(tmp, Wqb, qsb, nullptr, BL_, D_, D_, D_, D_, D_);
  cvt_k<<<8192, 256, 0, stream>>>(k, tmp, BL_ * D_, 1.f);
  gemm_k<1><<<gP, 256, 0, stream>>>(tmp, Wkb, ksb, nullptr, BL_, D_, D_, D_, D_, D_);
  cvt_k<<<8192, 256, 0, stream>>>(v, tmp, BL_ * D_, 1.f);
  gemm_k<2><<<gP, 256, 0, stream>>>(tmp, Wvb, vst, nullptr, BL_, D_, D_, D_, D_, D_);

  // 3. attn0 (streamed over key segments) + reduction/normalization
  attn0_k<<<B_ * H_ * NSEG, 256, 0, stream>>>(Ibf, ksb, vst, pHm, pRs);
  reduce0_k<<<B_ * H_, 256, 0, stream>>>(pHm, pRs, Hm0);

  // 4. interposed Ww projection on merged heads + split to per-head layouts
  gemm_k<0><<<dim3(B_ * NI_ / 64, D_ / 256), 256, 0, stream>>>(
      Hm0, Wwb, HmW, nullptr, B_ * NI_, D_, D_, D_, D_, D_);
  hmsplit_k<<<(B_ * NI_ * D_) / 256, 256, 0, stream>>>(HmW, Hmb, HmT);

  // 5. attn1 -> O (bf16, reuses tmp)
  attn1_k<<<B_ * (L_ / 64), 256, 0, stream>>>(qsb, Hmb, HmT, tmp);

  // 6. output projection + residual, then in-place LayerNorm on d_out
  gemm_k<3><<<gP, 256, 0, stream>>>(tmp, Wfcb, d_out, q, BL_, D_, D_, D_, D_, D_);
  ln_k<<<BL_, 256, 0, stream>>>((float*)d_out, lnw, lnb);
}
